// Agentembedding_18313740550795
// MI455X (gfx1250) — compile-verified, matrix operand order checked by reading, not run on hardware
//
#include <hip/hip_runtime.h>

typedef __attribute__((ext_vector_type(16))) __bf16 v16bf;
typedef __attribute__((ext_vector_type(8)))  float  v8f;
typedef __attribute__((ext_vector_type(4)))  unsigned int u32x4;
typedef __attribute__((ext_vector_type(8)))  int i32x8;
typedef __attribute__((ext_vector_type(4)))  int i32x4;

#define BB 32
#define QQ 256
#define NNODE 4096
#define DD 512
#define KDIM 512
#define VDIM 512
#define LDS_PITCH 68  // 64 elems + 4 (TDM pad: 2 DWORDs per 128B row) -> 136B pitch

static_assert(sizeof(v16bf) == 32, "v16bf must be 32 bytes");

#if __has_include(<hip/amd_detail/amd_gfx1250_TDM.h>)
#define TDM_6ARG 1
#endif

// ---------------------------------------------------------------------------
// WMMA helper (CDNA5 V_WMMA_F32_16X16X32_BF16, wave32)
// ---------------------------------------------------------------------------
__device__ __forceinline__ v8f wmma_bf16(v16bf a, v16bf b, v8f c) {
  return __builtin_amdgcn_wmma_f32_16x16x32_bf16(false, a, false, b, (short)0, c,
                                                 false, false);
}

// A/B fragment from row-major bf16 (B via transposed storage):
// lane l holds row (row0 + (l&15)), K chunks [off..off+7] and [off+16..off+23],
// off = (l>=16)?8:0 -> two contiguous 16-byte loads.
__device__ __forceinline__ v16bf ldfrag(const __bf16* __restrict__ base, int ld,
                                        int row0, int k0) {
  const int l = threadIdx.x & 31;
  const __bf16* p = base + (size_t)(row0 + (l & 15)) * ld + (k0 + ((l >> 4) << 3));
  union { uint4 q[2]; v16bf v; } u;
  u.q[0] = *(const uint4*)(p);
  u.q[1] = *(const uint4*)(p + 16);
  return u.v;
}

// Fragment from the LDS tile [128 rows][64 k], padded pitch LDS_PITCH elems.
__device__ __forceinline__ v16bf ldfrag_lds(const __bf16* tile, int row0, int ksub) {
  const int l = threadIdx.x & 31;
  const __bf16* p = tile + (row0 + (l & 15)) * LDS_PITCH + ksub + ((l >> 4) << 3);
  union { uint4 q[2]; v16bf v; } u;
  u.q[0] = *(const uint4*)(p);
  u.q[1] = *(const uint4*)(p + 16);
  return u.v;
}

// Fragment from fp32 source with on-the-fly bf16 conversion.
__device__ __forceinline__ v16bf ldfrag_f32(const float* __restrict__ base, int ld,
                                            int row0, int k0) {
  const int l = threadIdx.x & 31;
  const float* p = base + (size_t)(row0 + (l & 15)) * ld + (k0 + ((l >> 4) << 3));
  float4 f0 = *(const float4*)(p);
  float4 f1 = *(const float4*)(p + 4);
  float4 f2 = *(const float4*)(p + 16);
  float4 f3 = *(const float4*)(p + 20);
  v16bf v;
  v[0]  = (__bf16)f0.x; v[1]  = (__bf16)f0.y; v[2]  = (__bf16)f0.z; v[3]  = (__bf16)f0.w;
  v[4]  = (__bf16)f1.x; v[5]  = (__bf16)f1.y; v[6]  = (__bf16)f1.z; v[7]  = (__bf16)f1.w;
  v[8]  = (__bf16)f2.x; v[9]  = (__bf16)f2.y; v[10] = (__bf16)f2.z; v[11] = (__bf16)f2.w;
  v[12] = (__bf16)f3.x; v[13] = (__bf16)f3.y; v[14] = (__bf16)f3.z; v[15] = (__bf16)f3.w;
  return v;
}

// C/D fragment: lane l -> column n = col0+(l&15), rows m0+i, m0 = row0+8*(l>>4).
__device__ __forceinline__ void st_f32(float* __restrict__ C, int ldc, int row0,
                                       int col0, v8f acc) {
  const int l = threadIdx.x & 31;
  const int n = col0 + (l & 15);
  const int m0 = row0 + ((l >> 4) << 3);
#pragma unroll
  for (int i = 0; i < 8; ++i) C[(size_t)(m0 + i) * ldc + n] = acc[i];
}

__device__ __forceinline__ void st_bf16(__bf16* __restrict__ C, int ldc, int row0,
                                        int col0, v8f acc, const float* bias,
                                        float scale) {
  const int l = threadIdx.x & 31;
  const int n = col0 + (l & 15);
  const int m0 = row0 + ((l >> 4) << 3);
  const float bv = bias ? bias[n] : 0.0f;
#pragma unroll
  for (int i = 0; i < 8; ++i)
    C[(size_t)(m0 + i) * ldc + n] = (__bf16)(acc[i] * scale + bv);
}

// Transposed store Ct[n][m]: the 8 values per lane are contiguous -> one 16B store.
__device__ __forceinline__ void st_bf16_T(__bf16* __restrict__ Ct, int ldt, int row0,
                                          int col0, v8f acc, const float* bias) {
  const int l = threadIdx.x & 31;
  const int n = col0 + (l & 15);
  const int m0 = row0 + ((l >> 4) << 3);
  const float bv = bias[n];
  union { uint4 q; __bf16 h[8]; } u;
#pragma unroll
  for (int i = 0; i < 8; ++i) u.h[i] = (__bf16)(acc[i] + bv);
  *(uint4*)(Ct + (size_t)n * ldt + m0) = u.q;
}

// ---------------------------------------------------------------------------
// Tensor Data Mover: DMA a 2D bf16 tile [128 rows x 64 elems] from global
// (row stride stride_d0 elems) into LDS at lds_off.  LDS pad: +2 DWORDs per
// 128B row (pad_interval=32 DWORDs, pad_amount=2) -> 136B pitch, bank-friendly.
// D# layout per CDNA5 ISA ch.8.
// ---------------------------------------------------------------------------
__device__ __forceinline__ void tdm_load_tile(unsigned lds_off, const __bf16* gptr,
                                              int tensor_d0, int tensor_d1,
                                              int stride_d0) {
  unsigned long long ga = (unsigned long long)(uintptr_t)gptr;
  u32x4 g0;
  g0[0] = 1u;                                   // count=1, user descriptor
  g0[1] = lds_off;                              // lds_addr (bytes)
  g0[2] = (unsigned)(ga & 0xffffffffu);         // global_addr[31:0]
  g0[3] = (unsigned)((ga >> 32) & 0x01ffffffu)  // global_addr[56:32]
          | (2u << 30);                         // type = 2 ("image")
  i32x8 g1;
  g1[0] = 0x03110000;                           // data_size=2B, pad_en, int=32DW, amt=2DW
  g1[1] = (int)(((unsigned)tensor_d0 & 0xffffu) << 16);            // dim0[15:0]
  g1[2] = (int)(((unsigned)tensor_d0 >> 16) |
                (((unsigned)tensor_d1 & 0xffffu) << 16));          // dim0 hi, dim1 lo
  g1[3] = (int)(((unsigned)tensor_d1 >> 16) | (64u << 16));        // dim1 hi, tile_d0=64
  g1[4] = 128;                                  // tile_dim1=128, tile_dim2=0
  g1[5] = stride_d0;                            // tensor_dim0_stride[31:0]
  g1[6] = 0;
  g1[7] = 0;
  i32x4 z = {0, 0, 0, 0};
#if defined(TDM_6ARG)
  i32x8 z8 = {0, 0, 0, 0, 0, 0, 0, 0};
  __builtin_amdgcn_tensor_load_to_lds(g0, g1, z, z, z8, 0);
#else
  __builtin_amdgcn_tensor_load_to_lds(g0, g1, z, z, 0);
#endif
}

// ---------------------------------------------------------------------------
// Stage 0: weight convert + transpose  Wt[n][k] = (bf16)W[k][n]
// ---------------------------------------------------------------------------
__global__ void wconv_t(const float* __restrict__ W, __bf16* __restrict__ Wt,
                        int R, int C) {
  int i = blockIdx.x * blockDim.x + threadIdx.x;
  if (i >= R * C) return;
  int n = i / R;
  int k = i % R;
  Wt[(size_t)n * R + k] = (__bf16)W[(size_t)k * C + n];
}

// ---------------------------------------------------------------------------
// Stage 1: q = f_c @ Wq + bq  (fp32 A, bf16 Bt, bf16 out).
// Pipelined with two register sets (k-unroll by 2) -> no register rotation.
// ---------------------------------------------------------------------------
__global__ __launch_bounds__(256) void proj_q_kernel(
    const float* __restrict__ A, const __bf16* __restrict__ Bt,
    const float* __restrict__ bias, __bf16* __restrict__ C, int KA, int Ncols) {
  const int w = threadIdx.x >> 5;
  const int mw = w >> 1, nw = w & 1;
  const int row0 = blockIdx.y * 128 + mw * 32;
  const int col0 = blockIdx.x * 128 + nw * 64;
  v8f acc[2][4] = {};
  v16bf ea0 = ldfrag_f32(A, KA, row0, 0);
  v16bf ea1 = ldfrag_f32(A, KA, row0 + 16, 0);
  v16bf eb[4];
#pragma unroll
  for (int j = 0; j < 4; ++j) eb[j] = ldfrag(Bt, KA, col0 + 16 * j, 0);
#pragma unroll 1
  for (int k = 0; k < KA; k += 64) {
    // odd set loads (k+32 always < KA)
    v16bf oa0 = ldfrag_f32(A, KA, row0, k + 32);
    v16bf oa1 = ldfrag_f32(A, KA, row0 + 16, k + 32);
    v16bf ob[4];
#pragma unroll
    for (int j = 0; j < 4; ++j) ob[j] = ldfrag(Bt, KA, col0 + 16 * j, k + 32);
#pragma unroll
    for (int j = 0; j < 4; ++j) {
      acc[0][j] = wmma_bf16(ea0, eb[j], acc[0][j]);
      acc[1][j] = wmma_bf16(ea1, eb[j], acc[1][j]);
    }
    if (k + 64 < KA) {  // even set loads for next pair
      ea0 = ldfrag_f32(A, KA, row0, k + 64);
      ea1 = ldfrag_f32(A, KA, row0 + 16, k + 64);
#pragma unroll
      for (int j = 0; j < 4; ++j) eb[j] = ldfrag(Bt, KA, col0 + 16 * j, k + 64);
    }
#pragma unroll
    for (int j = 0; j < 4; ++j) {
      acc[0][j] = wmma_bf16(oa0, ob[j], acc[0][j]);
      acc[1][j] = wmma_bf16(oa1, ob[j], acc[1][j]);
    }
  }
#pragma unroll
  for (int i = 0; i < 2; ++i)
#pragma unroll
    for (int j = 0; j < 4; ++j)
      st_bf16(C, Ncols, row0 + 16 * i, col0 + 16 * j, acc[i][j], bias, 1.0f);
}

// ---------------------------------------------------------------------------
// Stage 2: k = f@Wk + bk (row-major), v = f@Wv + bv (stored transposed)
// ---------------------------------------------------------------------------
__global__ __launch_bounds__(256) void proj_kv_kernel(
    const float* __restrict__ f, const __bf16* __restrict__ Wkt,
    const __bf16* __restrict__ Wvt, const float* __restrict__ bk,
    const float* __restrict__ bv, __bf16* __restrict__ kb,
    __bf16* __restrict__ vbt) {
  const int w = threadIdx.x >> 5;
  const int mw = w >> 1, nw = w & 1;
  const int row0 = blockIdx.y * 128 + mw * 32;   // flat node index b*4096+n
  const int col0 = blockIdx.x * 128 + nw * 64;
  v8f acck[2][4] = {}, accv[2][4] = {};
  v16bf ea0 = ldfrag_f32(f, DD, row0, 0);
  v16bf ea1 = ldfrag_f32(f, DD, row0 + 16, 0);
  v16bf eK[4], eV[4];
#pragma unroll
  for (int j = 0; j < 4; ++j) {
    eK[j] = ldfrag(Wkt, DD, col0 + 16 * j, 0);
    eV[j] = ldfrag(Wvt, DD, col0 + 16 * j, 0);
  }
#pragma unroll 1
  for (int k = 0; k < DD; k += 64) {
    v16bf oa0 = ldfrag_f32(f, DD, row0, k + 32);
    v16bf oa1 = ldfrag_f32(f, DD, row0 + 16, k + 32);
    v16bf oK[4], oV[4];
#pragma unroll
    for (int j = 0; j < 4; ++j) {
      oK[j] = ldfrag(Wkt, DD, col0 + 16 * j, k + 32);
      oV[j] = ldfrag(Wvt, DD, col0 + 16 * j, k + 32);
    }
#pragma unroll
    for (int j = 0; j < 4; ++j) {
      acck[0][j] = wmma_bf16(ea0, eK[j], acck[0][j]);
      acck[1][j] = wmma_bf16(ea1, eK[j], acck[1][j]);
      accv[0][j] = wmma_bf16(ea0, eV[j], accv[0][j]);
      accv[1][j] = wmma_bf16(ea1, eV[j], accv[1][j]);
    }
    if (k + 64 < DD) {
      ea0 = ldfrag_f32(f, DD, row0, k + 64);
      ea1 = ldfrag_f32(f, DD, row0 + 16, k + 64);
#pragma unroll
      for (int j = 0; j < 4; ++j) {
        eK[j] = ldfrag(Wkt, DD, col0 + 16 * j, k + 64);
        eV[j] = ldfrag(Wvt, DD, col0 + 16 * j, k + 64);
      }
    }
#pragma unroll
    for (int j = 0; j < 4; ++j) {
      acck[0][j] = wmma_bf16(oa0, oK[j], acck[0][j]);
      acck[1][j] = wmma_bf16(oa1, oK[j], acck[1][j]);
      accv[0][j] = wmma_bf16(oa0, oV[j], accv[0][j]);
      accv[1][j] = wmma_bf16(oa1, oV[j], accv[1][j]);
    }
  }
  const int b = row0 >> 12;             // M tile never crosses batch (4096%128==0)
  const int nloc0 = row0 & (NNODE - 1);
  __bf16* vbt_b = vbt + (size_t)b * VDIM * NNODE;
#pragma unroll
  for (int i = 0; i < 2; ++i)
#pragma unroll
    for (int j = 0; j < 4; ++j) {
      st_bf16(kb, KDIM, row0 + 16 * i, col0 + 16 * j, acck[i][j], bk, 1.0f);
      st_bf16_T(vbt_b, NNODE, nloc0 + 16 * i, col0 + 16 * j, accv[i][j], bv);
    }
}

// ---------------------------------------------------------------------------
// TDM-staged GEMM body (stages 3/5): 128x128 C tile per WG.  Per 64-wide k
// step one TDM DMA stages the 128x64 B tile into LDS (ping/pong, padded
// pitch); loop is pair-unrolled so A fragments ping/pong between two named
// register sets with no copies.  STEPS = K/64, must be even.
// ---------------------------------------------------------------------------
#define TDM_GEMM_LOOP(Abase, Alda, Bt, Bld, Btd1, STEPS)                           \
  __shared__ __attribute__((aligned(128))) __bf16 btile[2][128 * LDS_PITCH];       \
  const int w = threadIdx.x >> 5;                                                  \
  const int mw = w >> 1, nw = w & 1;                                               \
  const int row0 = blockIdx.y * 128 + mw * 32;                                     \
  const int col0 = blockIdx.x * 128 + nw * 64;                                     \
  const __bf16* btbase = (Bt) + (size_t)(blockIdx.x * 128) * (Bld);                \
  if (w == 0)                                                                      \
    tdm_load_tile((unsigned)(uintptr_t)&btile[0][0], btbase, (Bld), (Btd1), (Bld)); \
  v8f acc[2][4] = {};                                                              \
  v16bf ea0 = ldfrag((Abase), (Alda), row0, 0);                                    \
  v16bf ea1 = ldfrag((Abase), (Alda), row0 + 16, 0);                               \
  v16bf ea2 = ldfrag((Abase), (Alda), row0, 32);                                   \
  v16bf ea3 = ldfrag((Abase), (Alda), row0 + 16, 32);                              \
  _Pragma("unroll 1")                                                              \
  for (int it = 0; it < (STEPS); it += 2) {                                        \
    /* ---- step it: LDS buf0, A even set ---- */                                  \
    if (w == 0) {                                                                  \
      tdm_load_tile((unsigned)(uintptr_t)&btile[1][0],                             \
                    btbase + (size_t)(it + 1) * 64, (Bld), (Btd1), (Bld));         \
      __builtin_amdgcn_s_wait_tensorcnt(1);                                        \
    }                                                                              \
    __syncthreads();                                                               \
    v16bf oa0 = ldfrag((Abase), (Alda), row0, (it + 1) * 64);                      \
    v16bf oa1 = ldfrag((Abase), (Alda), row0 + 16, (it + 1) * 64);                 \
    v16bf oa2 = ldfrag((Abase), (Alda), row0, (it + 1) * 64 + 32);                 \
    v16bf oa3 = ldfrag((Abase), (Alda), row0 + 16, (it + 1) * 64 + 32);            \
    _Pragma("unroll")                                                              \
    for (int j = 0; j < 4; ++j) {                                                  \
      v16bf bb = ldfrag_lds(&btile[0][0], nw * 64 + 16 * j, 0);                    \
      acc[0][j] = wmma_bf16(ea0, bb, acc[0][j]);                                   \
      acc[1][j] = wmma_bf16(ea1, bb, acc[1][j]);                                   \
    }                                                                              \
    _Pragma("unroll")                                                              \
    for (int j = 0; j < 4; ++j) {                                                  \
      v16bf bb = ldfrag_lds(&btile[0][0], nw * 64 + 16 * j, 32);                   \
      acc[0][j] = wmma_bf16(ea2, bb, acc[0][j]);                                   \
      acc[1][j] = wmma_bf16(ea3, bb, acc[1][j]);                                   \
    }                                                                              \
    __syncthreads();                                                               \
    /* ---- step it+1: LDS buf1, A odd set ---- */                                 \
    if (w == 0) {                                                                  \
      if (it + 2 < (STEPS)) {                                                      \
        tdm_load_tile((unsigned)(uintptr_t)&btile[0][0],                           \
                      btbase + (size_t)(it + 2) * 64, (Bld), (Btd1), (Bld));       \
        __builtin_amdgcn_s_wait_tensorcnt(1);                                      \
      } else {                                                                     \
        __builtin_amdgcn_s_wait_tensorcnt(0);                                      \
      }                                                                            \
    }                                                                              \
    __syncthreads();                                                               \
    if (it + 2 < (STEPS)) {                                                        \
      ea0 = ldfrag((Abase), (Alda), row0, (it + 2) * 64);                          \
      ea1 = ldfrag((Abase), (Alda), row0 + 16, (it + 2) * 64);                     \
      ea2 = ldfrag((Abase), (Alda), row0, (it + 2) * 64 + 32);                     \
      ea3 = ldfrag((Abase), (Alda), row0 + 16, (it + 2) * 64 + 32);                \
    }                                                                              \
    _Pragma("unroll")                                                              \
    for (int j = 0; j < 4; ++j) {                                                  \
      v16bf bb = ldfrag_lds(&btile[1][0], nw * 64 + 16 * j, 0);                    \
      acc[0][j] = wmma_bf16(oa0, bb, acc[0][j]);                                   \
      acc[1][j] = wmma_bf16(oa1, bb, acc[1][j]);                                   \
    }                                                                              \
    _Pragma("unroll")                                                              \
    for (int j = 0; j < 4; ++j) {                                                  \
      v16bf bb = ldfrag_lds(&btile[1][0], nw * 64 + 16 * j, 32);                   \
      acc[0][j] = wmma_bf16(oa2, bb, acc[0][j]);                                   \
      acc[1][j] = wmma_bf16(oa3, bb, acc[1][j]);                                   \
    }                                                                              \
    __syncthreads();                                                               \
  }

// ---------------------------------------------------------------------------
// Stage 3: S[b] = (q[b] @ k[b]^T) * scale  -> bf16 [Q, N]
// ---------------------------------------------------------------------------
__global__ __launch_bounds__(256) void score_kernel(
    const __bf16* __restrict__ qb, const __bf16* __restrict__ kbm,
    __bf16* __restrict__ S, float scale) {
  const int b = blockIdx.z;
  const __bf16* A = qb + (size_t)b * QQ * KDIM;
  const __bf16* Bt = kbm + (size_t)b * NNODE * KDIM;
  __bf16* Sb = S + (size_t)b * QQ * NNODE;
  TDM_GEMM_LOOP(A, KDIM, Bt, KDIM, NNODE, KDIM / 64)
#pragma unroll
  for (int i = 0; i < 2; ++i)
#pragma unroll
    for (int j = 0; j < 4; ++j)
      st_bf16(Sb, NNODE, row0 + 16 * i, col0 + 16 * j, acc[i][j], nullptr, scale);
}

// ---------------------------------------------------------------------------
// Stage 4: softmax over N per (b,q) row, in place (same footprint -> safe)
// ---------------------------------------------------------------------------
__global__ __launch_bounds__(256) void softmax_kernel(__bf16* __restrict__ S) {
  __shared__ float red[256];
  __bf16* p = S + (size_t)blockIdx.x * NNODE;
  const int t = threadIdx.x;
  float v[16];
  float mx = -3.0e38f;
#pragma unroll
  for (int i = 0; i < 16; ++i) {
    v[i] = (float)p[t * 16 + i];
    mx = fmaxf(mx, v[i]);
  }
  red[t] = mx;
  __syncthreads();
  for (int s = 128; s > 0; s >>= 1) {
    if (t < s) red[t] = fmaxf(red[t], red[t + s]);
    __syncthreads();
  }
  mx = red[0];
  __syncthreads();
  float sum = 0.0f;
#pragma unroll
  for (int i = 0; i < 16; ++i) {
    v[i] = __expf(v[i] - mx);
    sum += v[i];
  }
  red[t] = sum;
  __syncthreads();
  for (int s = 128; s > 0; s >>= 1) {
    if (t < s) red[t] += red[t + s];
    __syncthreads();
  }
  const float inv = 1.0f / red[0];
#pragma unroll
  for (int i = 0; i < 16; ++i) p[t * 16 + i] = (__bf16)(v[i] * inv);
}

// ---------------------------------------------------------------------------
// Stage 5: out[b] = P[b] @ v[b]  (P bf16 [Q,N], vbt bf16 [V,N], f32 out)
// ---------------------------------------------------------------------------
__global__ __launch_bounds__(256) void out_kernel(
    const __bf16* __restrict__ P, const __bf16* __restrict__ vbt,
    float* __restrict__ out) {
  const int b = blockIdx.z;
  const __bf16* A = P + (size_t)b * QQ * NNODE;
  const __bf16* Bt = vbt + (size_t)b * VDIM * NNODE;
  float* Cb = out + (size_t)b * QQ * VDIM;
  TDM_GEMM_LOOP(A, NNODE, Bt, NNODE, VDIM, NNODE / 64)
#pragma unroll
  for (int i = 0; i < 2; ++i)
#pragma unroll
    for (int j = 0; j < 4; ++j)
      st_f32(Cb, VDIM, row0 + 16 * i, col0 + 16 * j, acc[i][j]);
}

// ---------------------------------------------------------------------------
// Host launcher
// ---------------------------------------------------------------------------
extern "C" void kernel_launch(void* const* d_in, const int* in_sizes, int n_in,
                              void* d_out, int out_size, void* d_ws, size_t ws_size,
                              hipStream_t stream) {
  (void)in_sizes; (void)n_in; (void)out_size; (void)ws_size;
  const float* f_c = (const float*)d_in[0];
  const float* f   = (const float*)d_in[1];
  const float* Wq  = (const float*)d_in[2];
  const float* bq  = (const float*)d_in[3];
  const float* Wk  = (const float*)d_in[4];
  const float* bk  = (const float*)d_in[5];
  const float* Wv  = (const float*)d_in[6];
  const float* bv  = (const float*)d_in[7];
  float* out = (float*)d_out;

  char* ws = (char*)d_ws;
  size_t off = 0;
  auto alloc = [&](size_t bytes) -> void* {
    off = (off + 255) & ~(size_t)255;
    void* p = ws + off;
    off += bytes;
    return p;
  };
  __bf16* qb  = (__bf16*)alloc((size_t)BB * QQ * KDIM * 2);     //   8 MB
  __bf16* kb  = (__bf16*)alloc((size_t)BB * NNODE * KDIM * 2);  // 134 MB
  __bf16* vbt = (__bf16*)alloc((size_t)BB * VDIM * NNODE * 2);  // 134 MB
  __bf16* S   = (__bf16*)alloc((size_t)BB * QQ * NNODE * 2);    //  67 MB
  __bf16* Wqt = (__bf16*)alloc((size_t)2 * DD * KDIM * 2);
  __bf16* Wkt = (__bf16*)alloc((size_t)DD * KDIM * 2);
  __bf16* Wvt = (__bf16*)alloc((size_t)DD * VDIM * 2);

  wconv_t<<<dim3((2 * DD * KDIM + 255) / 256), 256, 0, stream>>>(Wq, Wqt, 2 * DD, KDIM);
  wconv_t<<<dim3((DD * KDIM + 255) / 256), 256, 0, stream>>>(Wk, Wkt, DD, KDIM);
  wconv_t<<<dim3((DD * VDIM + 255) / 256), 256, 0, stream>>>(Wv, Wvt, DD, VDIM);

  proj_q_kernel<<<dim3(KDIM / 128, (BB * QQ) / 128), 256, 0, stream>>>(
      f_c, Wqt, bq, qb, 2 * DD, KDIM);

  proj_kv_kernel<<<dim3(KDIM / 128, (BB * NNODE) / 128), 256, 0, stream>>>(
      f, Wkt, Wvt, bk, bv, kb, vbt);

  const float scale = 0.04419417382415922f;  // 1/sqrt(512)
  score_kernel<<<dim3(NNODE / 128, QQ / 128, BB), 256, 0, stream>>>(qb, kb, S, scale);

  softmax_kernel<<<dim3(BB * QQ), 256, 0, stream>>>(S);

  out_kernel<<<dim3(VDIM / 128, QQ / 128, BB), 256, 0, stream>>>(S, vbt, out);
}
